// modelLSTM_Kuai_22445499089341
// MI455X (gfx1250) — compile-verified, matrix operand order checked by reading, not run on hardware
//
#include <hip/hip_runtime.h>
#include <hip/hip_bf16.h>

// ---------------------------------------------------------------------------
// LSTM (T=365, N=2000, NX=16, H=256, NY=1) for gfx1250 (MI455X, wave32, WMMA)
//
// Rows are independent LSTMs. Block = 16 rows, 8 waves; each wave owns 128 of
// the 1024 gate columns. Per step: uniform 9-K-tile V_WMMA_F32_16X16X32_F16
// GEMM over [x_t | 1 | h] @ [wx ; b ; wh] (bias folded in via a constant-1
// A slot in the zero-padded x K-tile). B-fragments are double-buffered, and a
// sched_barrier pins the kt+1 prefetch ABOVE the kt WMMA group so the waits
// become loadcnt<=16 rather than 0 (L2 latency hidden behind the WMMA chain).
// h/c state is block-resident for all 365 steps; packed weights stay L2-hot.
// ---------------------------------------------------------------------------

typedef __attribute__((ext_vector_type(16))) _Float16 v16h;
typedef __attribute__((ext_vector_type(8)))  float    v8f;

#define LSTM_T 365
#define LSTM_N 2000
#define LSTM_NX 16
#define LSTM_H 256
#define NKT 9                       // kt0 = [x|1|0] tile, kt1..8 = h tiles
#define WHP_ELEMS (64 * NKT * 512)  // 64 ntiles * 9 ktiles * 512 f16/frag
#define GSTRIDE 1044                // f16 stride for gate buffer (bank-stagger)

#if defined(__has_builtin) && __has_builtin(__builtin_amdgcn_sched_barrier)
#define SCHED_FENCE() __builtin_amdgcn_sched_barrier(0)
#else
#define SCHED_FENCE() asm volatile("" ::: "memory")
#endif

// ----------------------------- weight repack -------------------------------
// B-fragment layout (16-bit WMMA operand striping, lane-contiguous 32B):
//   frag(nt, kt) at (nt*9+kt)*512 halves; element = lane*16 + j
//   kmap = (j&7) + (j>=8 ? 16 : 0) + (lane>=16 ? 8 : 0)
//   col  = nt*16 + (lane&15);  g = col>>8; hh = col&255
//   kt==0 : k = kmap       -> k<16: wx[g][k][hh]; k==16: b[col]; else 0
//   kt>=1 : k = kmap + (kt-1)*32 -> wh[g][k][hh]
__global__ __launch_bounds__(256) void lstm_pack_weights(
    const float* __restrict__ wx,   // [4][16][256]
    const float* __restrict__ wh,   // [4][256][256]
    const float* __restrict__ b,    // [4][256] flat
    _Float16* __restrict__ whp)     // [WHP_ELEMS]
{
    int idx = blockIdx.x * 256 + threadIdx.x;
    if (idx >= WHP_ELEMS) return;
    int frag = idx >> 9;
    int within = idx & 511;
    int lane = within >> 4;
    int j = within & 15;
    int nt = frag / NKT, kt = frag % NKT;
    int kmap = (j & 7) + ((j >= 8) ? 16 : 0) + ((lane >= 16) ? 8 : 0);
    int col = nt * 16 + (lane & 15);
    int g = col >> 8, hh = col & 255;
    float v;
    if (kt == 0) {
        if (kmap < LSTM_NX)       v = wx[(g * LSTM_NX + kmap) * 256 + hh];
        else if (kmap == LSTM_NX) v = b[col];       // bias row (A slot == 1.0)
        else                      v = 0.0f;
    } else {
        int k = kmap + (kt - 1) * 32;
        v = wh[(g * 256 + k) * 256 + hh];
    }
    whp[idx] = (_Float16)v;
}

// ------------------------------ activations --------------------------------
__device__ __forceinline__ float fsigmoid(float v) {
    return 1.0f / (1.0f + __expf(-v));          // v_exp_f32 + rcp
}
__device__ __forceinline__ float ftanh_(float v) {
    float e = __expf(-2.0f * v);
    return (1.0f - e) / (1.0f + e);
}

// ------------------------------- main kernel -------------------------------
__global__ __launch_bounds__(256) void lstm_persistent_kernel(
    const float* __restrict__ x,      // [T][N][NX]
    const float* __restrict__ w_lin,  // [H][1]
    const float* __restrict__ b_lin,  // [1]
    const _Float16* __restrict__ whp, // packed [wx;b;wh] fragments (L2-hot)
    float* __restrict__ out)          // [T][N][1]
{
    __shared__ __align__(16) _Float16 g_buf[16 * GSTRIDE];  // gate preacts (f16)
    __shared__ __align__(32) _Float16 a_buf[NKT * 512];     // [x|1|h] A-frags
    __shared__ float red_buf[256];

    const int tid  = threadIdx.x;
    const int lane = tid & 31;
    const int wv   = tid >> 5;            // 8 waves: cols [wv*128, wv*128+128)
    const int row0 = blockIdx.x * 16;

    // Zero all A fragments (h0 = 0, x padding = 0), then plant the constant-1
    // at k=16 of the kt0 tile (ko=16 -> laneHi=0, j=8 -> slot lane*16+8).
    for (int i = tid; i < NKT * 512; i += 256) a_buf[i] = (_Float16)0.0f;
    __syncthreads();
    if (tid < 16) a_buf[tid * 16 + 8] = (_Float16)1.0f;

    // Output head weights, per-thread slice: row r = tid>>4, hh = (tid&15)+16q
    float wl[16];
#pragma unroll
    for (int q = 0; q < 16; ++q)
        wl[q] = w_lin[(tid & 15) + 16 * q];
    const float blin = b_lin[0];

    // Per-thread cell state c[r][hh], r = tid>>4, hh = (tid&15)+16q
    float cst[16];
#pragma unroll
    for (int q = 0; q < 16; ++q) cst[q] = 0.0f;

    __syncthreads();

    for (int t = 0; t < LSTM_T; ++t) {
        // Opaque zero offset: blocks LICM of the B-fragment loads (which would
        // otherwise hoist ~640 VGPRs of data and spill to scratch), while the
        // addrspace(1) pointer keeps them GLOBAL (LOADcnt-only) loads.
        size_t off0 = 0;
        asm volatile("" : "+s"(off0));
        const _Float16* __restrict__ whp_t = whp + off0;

        // ---- stage x_t into kt0 A-fragment (k 0..15; k=16 stays 1.0) ----
        {
            int r = tid >> 4, i = tid & 15;
            float xv = x[((size_t)t * LSTM_N + row0 + r) * LSTM_NX + i];
            int laneHi = (i >> 3) & 1;
            int j = i & 7;
            a_buf[(r + (laneHi << 4)) * 16 + j] = (_Float16)xv;
        }
        __syncthreads();

        // ---- WMMA: g = [x|1|h] @ [wx;b;wh], 9 K-tiles, B double-buffered ----
        v8f acc[8];
#pragma unroll
        for (int nt = 0; nt < 8; ++nt)
#pragma unroll
            for (int e = 0; e < 8; ++e) acc[nt][e] = 0.0f;

        v16h bcur[8], bnext[8];
#pragma unroll
        for (int nt = 0; nt < 8; ++nt)
            bcur[nt] = *(const v16h*)(whp_t + (size_t)((wv * 8 + nt) * NKT) * 512 + lane * 16);

#pragma unroll
        for (int kt = 0; kt < NKT; ++kt) {
            // A fragment for this tile + prefetch of next tile's B fragments
            v16h afrag = *(const v16h*)(&a_buf[kt * 512 + lane * 16]);
            if (kt < NKT - 1) {
#pragma unroll
                for (int nt = 0; nt < 8; ++nt)
                    bnext[nt] = *(const v16h*)(whp_t +
                        (size_t)((wv * 8 + nt) * NKT + kt + 1) * 512 + lane * 16);
            }
            // Pin the prefetch above the WMMA group: the scheduler may not
            // sink these loads next to their uses, so the wait before the
            // WMMAs is loadcnt<=16 (prefetch in flight), not loadcnt==0.
            SCHED_FENCE();
#pragma unroll
            for (int nt = 0; nt < 8; ++nt) {
                acc[nt] = __builtin_amdgcn_wmma_f32_16x16x32_f16(
                    false, afrag, false, bcur[nt], (short)0, acc[nt], false, false);
            }
            SCHED_FENCE();
#pragma unroll
            for (int nt = 0; nt < 8; ++nt) bcur[nt] = bnext[nt];
        }

        // ---- spill gate preacts to LDS (C layout: row = v + (lane>=16)*8) ----
        {
            int rbase = (lane >= 16) ? 8 : 0;
#pragma unroll
            for (int nt = 0; nt < 8; ++nt) {
                int col = wv * 128 + nt * 16 + (lane & 15);
#pragma unroll
                for (int v = 0; v < 8; ++v)
                    g_buf[(v + rbase) * GSTRIDE + col] = (_Float16)acc[nt][v];
            }
        }
        __syncthreads();

        // ---- elementwise: gates -> c,h update; write h back as A-fragments ----
        {
            int r = tid >> 4, hb = tid & 15;
            float partial = 0.0f;
#pragma unroll
            for (int q = 0; q < 16; ++q) {
                int hh = hb + q * 16;
                float gi = (float)g_buf[r * GSTRIDE + hh];
                float gf = (float)g_buf[r * GSTRIDE + 256 + hh];
                float gc = (float)g_buf[r * GSTRIDE + 512 + hh];
                float go = (float)g_buf[r * GSTRIDE + 768 + hh];
                float iv = fsigmoid(gi);
                float fv = fsigmoid(gf);
                float gt = ftanh_(gc);
                float ov = fsigmoid(go);
                float cn = fv * cst[q] + iv * gt;
                cst[q] = cn;
                float hv = ov * ftanh_(cn);
                partial += hv * wl[q];
                // scatter h into next step's A-fragment slot (tiles 1..8)
                int kt = (hh >> 5) + 1, ko = hh & 31;
                int laneHi = (ko >> 3) & 1;
                int j = (ko & 7) + ((ko >= 16) ? 8 : 0);
                a_buf[kt * 512 + (r + laneHi * 16) * 16 + j] = (_Float16)hv;
            }
            red_buf[tid] = partial;
        }
        __syncthreads();

        // ---- output head: out[t][row] = h . w_lin + b_lin ----
        if (tid < 16) {
            float s = blin;
#pragma unroll
            for (int m = 0; m < 16; ++m) s += red_buf[tid * 16 + m];
            out[(size_t)t * LSTM_N + row0 + tid] = s;
        }
        // red_buf readers reach the next barrier before any rewrite (2 barriers
        // away), so no extra loop-end barrier is needed.
    }
}

// ------------------------------- launcher ----------------------------------
extern "C" void kernel_launch(void* const* d_in, const int* in_sizes, int n_in,
                              void* d_out, int out_size, void* d_ws, size_t ws_size,
                              hipStream_t stream) {
    (void)in_sizes; (void)n_in; (void)out_size; (void)ws_size;
    const float* x     = (const float*)d_in[0];   // [365,2000,16]
    const float* wx    = (const float*)d_in[1];   // [4,16,256]
    const float* wh    = (const float*)d_in[2];   // [4,256,256]
    const float* b     = (const float*)d_in[3];   // [4,256]
    const float* w_lin = (const float*)d_in[4];   // [256,1]
    const float* b_lin = (const float*)d_in[5];   // [1]
    float* out = (float*)d_out;                   // [365,2000,1]

    _Float16* whp = (_Float16*)d_ws;              // 576 KB packed [wx;b;wh]

    {   // repack weights (+bias row) into WMMA-B fragment layout (f16)
        int blocks = (WHP_ELEMS + 255) / 256;     // 1152
        lstm_pack_weights<<<blocks, 256, 0, stream>>>(wx, wh, b, whp);
    }

    // 2000 rows / 16 rows per block = 125 persistent blocks
    lstm_persistent_kernel<<<LSTM_N / 16, 256, 0, stream>>>(
        x, w_lin, b_lin, whp, out);
}
